// Transition_61289183314348
// MI455X (gfx1250) — compile-verified, matrix operand order checked by reading, not run on hardware
//
#include <hip/hip_runtime.h>
#include <stdint.h>

// ---------------------------------------------------------------------------
// LSTM scan for MI455X (gfx1250, wave32, WMMA + TDM).
//
//   T=2048 steps, B=256, D_IN=16, HID=256, GATES=1024, OUT=3+3.
//
//   * pack_weights: one-shot f32->bf16 conversion of W_hh / W_ih into the
//     exact WMMA B-fragment lane layout (32 contiguous bytes per lane per
//     16x32 K-tile) in d_ws. W_hh bf16 = 512KB -> L2-resident (192MB L2),
//     re-streamed every step with zero per-step conversion cost.
//   * lstm_scan: 16 WGs (one 16-row batch tile), 512 threads = 16 waves,
//     each wave owns 64 gate columns (4 N-tiles). Per step: 4 x-projection
//     WMMAs + 32 recurrent WMMAs (v_wmma_f32_16x16x32_bf16, f32 accum
//     seeded with b_ih+b_hh). Weight fragments double-buffered in registers
//     (memory clobber stops LICM from hoisting 256 VGPRs of weights and
//     spilling them to scratch, which round 1 showed). h kept in LDS bf16,
//     gates staged in LDS f32, OUT heads fused via ds_add_f32.
//   * TDM: one tensor_load_to_lds per step prefetches the next step's
//     16x16 f32 x-tile into double-buffered LDS (all 16 waves share it),
//     synchronized with s_wait_tensorcnt + the existing barrier.
// ---------------------------------------------------------------------------

#define T_STEPS 2048
#define BATCH   256
#define DIN     16
#define HID     256
#define OUTD    6

#define BTILE   16
#define NBLOCKS (BATCH / BTILE)   // 16
#define NTHREADS 512              // 16 wave32 waves
#define NT_PER_WAVE 4             // 16-col N-tiles per wave (64 cols)
#define KTILES  8                 // 256 / 32

#define GROW 1028                 // gates LDS row stride (f32), padded
#define HROW 132                  // h LDS row stride (u32 = 2 bf16), padded

#define WHH_PK_U32 (64 * 8 * 32 * 8)   // [ntile][ktile][lane][8 u32]
#define WIH_PK_U32 (64 * 32 * 8)       // [ntile][lane][8 u32]

#if defined(__has_builtin)
#if __has_builtin(__builtin_amdgcn_tensor_load_to_lds) && \
    __has_builtin(__builtin_amdgcn_s_wait_tensorcnt)
#define HAVE_TDM 1
#endif
#endif
#ifndef HAVE_TDM
#define HAVE_TDM 0
#endif

typedef __bf16 v16bf __attribute__((ext_vector_type(16)));
typedef float  v8f   __attribute__((ext_vector_type(8)));

union FragU { unsigned u[8]; uint4 q[2]; v16bf v; };

__device__ __forceinline__ unsigned pk_bf16(float a, float b) {
  unsigned ua = __float_as_uint(a);
  unsigned ub = __float_as_uint(b);
  ua = (ua + 0x7FFFu + ((ua >> 16) & 1u)) >> 16;          // rne -> low half
  ub = (ub + 0x7FFFu + ((ub >> 16) & 1u)) & 0xFFFF0000u;  // rne -> high half
  return ua | ub;
}

__device__ __forceinline__ float fast_sig(float x) {
  return 1.0f / (1.0f + __expf(-x));
}
__device__ __forceinline__ float fast_tanh(float x) {
  return 1.0f - 2.0f / (__expf(2.0f * x) + 1.0f);
}

#if HAVE_TDM
typedef unsigned u32x4 __attribute__((ext_vector_type(4)));
typedef int      i32x4 __attribute__((ext_vector_type(4)));
typedef int      i32x8 __attribute__((ext_vector_type(8)));

// TDM: 2D tile load, 16 rows x 16 f32 (one x-timestep for this batch tile),
// viewing xs as a (T*B) x 16 f32 tensor with row stride 16.
__device__ __forceinline__ void tdm_load_x_tile(const float* gsrc,
                                                unsigned lds_byte_off) {
  unsigned long long ga = (unsigned long long)(uintptr_t)gsrc;
  u32x4 g0;
  g0[0] = 1u;                                              // count=1 (valid)
  g0[1] = lds_byte_off;                                    // LDS dest
  g0[2] = (unsigned)(ga & 0xFFFFFFFFu);                    // gaddr[31:0]
  g0[3] = (unsigned)((ga >> 32) & 0x1FFFFFFu) | (2u << 30); // gaddr[56:32]|type=2
  i32x8 g1;
  g1[0] = (int)(2u << 16);            // data_size=4B, no multicast
  g1[1] = (int)(16u << 16);           // tensor_dim0 = 16 (lo16)
  g1[2] = 0;                          // dim0 hi=0, dim1 lo16 = 0 (524288)
  g1[3] = (int)((16u << 16) | 0x8u);  // dim1 hi16 = 8, tile_dim0 = 16
  g1[4] = 16;                         // tile_dim1 = 16, tile_dim2 = 0
  g1[5] = 16;                         // tensor_dim0_stride = 16
  g1[6] = 0;
  g1[7] = 0;
  i32x4 z4 = {0, 0, 0, 0};
#if __clang_major__ >= 23
  i32x8 z8 = {0, 0, 0, 0, 0, 0, 0, 0};
  __builtin_amdgcn_tensor_load_to_lds(g0, g1, z4, z4, z8, 0);
#else
  __builtin_amdgcn_tensor_load_to_lds(g0, g1, z4, z4, 0);
#endif
}
#endif

// ---------------------------------------------------------------------------
// Pack W_hh (1024x256 f32) and W_ih (1024x16 f32, K zero-padded to 32) into
// bf16 WMMA B-fragment layout.
// ---------------------------------------------------------------------------
__global__ void pack_weights(const float* __restrict__ Wih,
                             const float* __restrict__ Whh,
                             unsigned* __restrict__ whh_pk,
                             unsigned* __restrict__ wih_pk) {
  int tid = blockIdx.x * blockDim.x + threadIdx.x;
  if (tid < WHH_PK_U32) {
    int j    = tid & 7;
    int lane = (tid >> 3) & 31;
    int kt   = (tid >> 8) & 7;
    int nt   = tid >> 11;
    int g  = nt * 16 + (lane & 15);
    int k0 = kt * 32 + ((lane < 16) ? 0 : 8);
    int k  = k0 + ((j < 4) ? (j * 2) : (16 + (j - 4) * 2));
    whh_pk[tid] = pk_bf16(Whh[g * HID + k], Whh[g * HID + k + 1]);
  }
  int tid2 = tid - WHH_PK_U32;
  if (tid2 >= 0 && tid2 < WIH_PK_U32) {
    int j    = tid2 & 7;
    int lane = (tid2 >> 3) & 31;
    int nt   = tid2 >> 8;
    unsigned val = 0u;
    if (j < 4) {  // K 16..31 is zero padding
      int g = nt * 16 + (lane & 15);
      int k = ((lane < 16) ? 0 : 8) + j * 2;
      val = pk_bf16(Wih[g * DIN + k], Wih[g * DIN + k + 1]);
    }
    wih_pk[tid2] = val;
  }
}

// ---------------------------------------------------------------------------
// Persistent per-batch-tile LSTM scan.
// ---------------------------------------------------------------------------
__global__ __launch_bounds__(NTHREADS, 1)
void lstm_scan(const float* __restrict__ xs,
               const unsigned* __restrict__ whh_pk,
               const unsigned* __restrict__ wih_pk,
               const float* __restrict__ b_ih,
               const float* __restrict__ b_hh,
               const float* __restrict__ Wuvw,
               const float* __restrict__ buvw,
               const float* __restrict__ Wpqr,
               const float* __restrict__ bpqr,
               float* __restrict__ out) {
  __shared__ __align__(16) float    gates[BTILE * GROW];    // ~64.3 KB
  __shared__ __align__(16) unsigned h_lds[BTILE * HROW];    // ~8.3 KB bf16 h
  __shared__ __align__(16) float    xstage[2][BTILE][DIN];  // 2 KB (TDM dst)
  __shared__ float out_stage[BTILE][8];

  const int tid  = threadIdx.x;
  const int lane = tid & 31;
  const int wave = tid >> 5;
  const int b0   = blockIdx.x * BTILE;

  // --- WMMA-phase lane constants
  const int arow    = lane & 15;             // A row (batch within tile)
  const int asel    = (lane < 16) ? 0 : 4;   // u32 select within K-tile
  const int xkoff   = (lane < 16) ? 0 : 8;   // x fragment K offset
  const int colbase = wave * (NT_PER_WAVE * 16) + (lane & 15);
  const int rbase   = (lane < 16) ? 0 : 8;   // C/D row base for this lane

  float bias[NT_PER_WAVE];
#pragma unroll
  for (int nt = 0; nt < NT_PER_WAVE; ++nt) {
    int g = colbase + nt * 16;
    bias[nt] = b_ih[g] + b_hh[g];
  }

  // --- elementwise-phase thread constants: (row, 8 consecutive hidden j)
  const int erow = tid & 15;
  const int ej0  = (tid >> 4) * 8;  // 0..248
  float wu[3][8], wp[3][8];
#pragma unroll
  for (int k = 0; k < 3; ++k) {
#pragma unroll
    for (int jj = 0; jj < 8; ++jj) {
      wu[k][jj] = Wuvw[k * HID + ej0 + jj];
      wp[k][jj] = Wpqr[k * HID + ej0 + jj];
    }
  }
  float c[8];
#pragma unroll
  for (int jj = 0; jj < 8; ++jj) c[jj] = 0.0f;

  // zero h (h0 = 0) and output staging; fetch x[0] tile
  for (int i = tid; i < BTILE * HROW; i += NTHREADS) h_lds[i] = 0u;
  if (tid < 128) out_stage[tid >> 3][tid & 7] = 0.0f;
#if HAVE_TDM
  if (tid == 0) {
    tdm_load_x_tile(xs + (size_t)b0 * DIN,
                    (unsigned)(uintptr_t)&xstage[0][0][0]);
    __builtin_amdgcn_s_wait_tensorcnt((short)0);
  }
#endif
  __syncthreads();

  // this wave's packed W_hh fragments: base + (nt*KTILES + kt)*256 u32
  const unsigned* whh_w = whh_pk + (size_t)(wave * NT_PER_WAVE) * KTILES * 256
                                 + (size_t)lane * 8;

  // W_ih fragments: tiny (4 x v16bf), register resident across all steps
  FragU fih[NT_PER_WAVE];
#pragma unroll
  for (int nt = 0; nt < NT_PER_WAVE; ++nt) {
    const uint4* p = (const uint4*)(wih_pk +
        ((size_t)(wave * NT_PER_WAVE + nt) * 32 + lane) * 8);
    fih[nt].q[0] = p[0];
    fih[nt].q[1] = p[1];
  }

#pragma unroll 1
  for (int t = 0; t < T_STEPS; ++t) {
    // Opaque memory clobber: stops LICM from hoisting (and then spilling)
    // the per-step W_hh fragment stream out of the t-loop. Values already
    // in registers (fih, bias, wu, wp) are unaffected.
    asm volatile("" ::: "memory");

#if HAVE_TDM
    // kick off the TDM prefetch of the next step's x-tile (other LDS buffer)
    if (t + 1 < T_STEPS && tid == 0) {
      tdm_load_x_tile(xs + ((size_t)(t + 1) * BATCH + b0) * DIN,
                      (unsigned)(uintptr_t)&xstage[(t + 1) & 1][0][0]);
    }
#endif

    // ---- x A-fragment (f32 -> bf16, K padded 16 -> 32 with zeros)
#if HAVE_TDM
    const float* xrow = &xstage[t & 1][arow][xkoff];
    float4 x0 = *(const float4*)xrow;
    float4 x1 = *(const float4*)(xrow + 4);
#else
    const float* xp = xs + ((size_t)t * BATCH + b0 + arow) * DIN + xkoff;
    float4 x0 = *(const float4*)xp;
    float4 x1 = *(const float4*)(xp + 4);
    if (t + 1 < T_STEPS)
      __builtin_prefetch(xs + ((size_t)(t + 1) * BATCH + b0 + arow) * DIN, 0, 1);
#endif
    FragU xa;
    xa.u[0] = pk_bf16(x0.x, x0.y);
    xa.u[1] = pk_bf16(x0.z, x0.w);
    xa.u[2] = pk_bf16(x1.x, x1.y);
    xa.u[3] = pk_bf16(x1.z, x1.w);
    xa.u[4] = 0u; xa.u[5] = 0u; xa.u[6] = 0u; xa.u[7] = 0u;

    // ---- gate accumulators: bias + x @ W_ih^T
    v8f acc[NT_PER_WAVE];
#pragma unroll
    for (int nt = 0; nt < NT_PER_WAVE; ++nt) {
      v8f a;
#pragma unroll
      for (int r = 0; r < 8; ++r) a[r] = bias[nt];
      acc[nt] = __builtin_amdgcn_wmma_f32_16x16x32_bf16(
          false, xa.v, false, fih[nt].v, (short)0, a, false, false);
    }

    // ---- recurrent term: += h @ W_hh^T.
    // Weight fragments double-buffered in registers: K-tile kt+1 streams
    // from L2 while the 4 WMMAs of K-tile kt execute.
    FragU wfr[2][NT_PER_WAVE];
#pragma unroll
    for (int nt = 0; nt < NT_PER_WAVE; ++nt) {
      const uint4* w4 = (const uint4*)(whh_w + (nt * KTILES) * 256);
      wfr[0][nt].q[0] = w4[0];
      wfr[0][nt].q[1] = w4[1];
    }
#pragma unroll
    for (int kt = 0; kt < KTILES; ++kt) {
      const int cur = kt & 1;
      const int nxt = cur ^ 1;
      if (kt + 1 < KTILES) {
#pragma unroll
        for (int nt = 0; nt < NT_PER_WAVE; ++nt) {
          const uint4* w4 = (const uint4*)(whh_w + (nt * KTILES + kt + 1) * 256);
          wfr[nxt][nt].q[0] = w4[0];
          wfr[nxt][nt].q[1] = w4[1];
        }
      }
      FragU ha;
      const unsigned* hb = &h_lds[arow * HROW + kt * 16 + asel];
      ha.q[0] = *(const uint4*)(hb);
      ha.q[1] = *(const uint4*)(hb + 8);
#pragma unroll
      for (int nt = 0; nt < NT_PER_WAVE; ++nt) {
        acc[nt] = __builtin_amdgcn_wmma_f32_16x16x32_bf16(
            false, ha.v, false, wfr[cur][nt].v, (short)0, acc[nt], false, false);
      }
    }

    // ---- stage pre-activation gates to LDS
#pragma unroll
    for (int nt = 0; nt < NT_PER_WAVE; ++nt) {
#pragma unroll
      for (int r = 0; r < 8; ++r) {
        gates[(rbase + r) * GROW + colbase + nt * 16] = acc[nt][r];
      }
    }
    __syncthreads();

    // ---- elementwise LSTM cell + fused OUT-head partials
    union { float4 q[2]; float f[8]; } gi, gf, gg, go;
    const float* grow = &gates[erow * GROW + ej0];
    gi.q[0] = *(const float4*)(grow + 0 * HID);
    gi.q[1] = *(const float4*)(grow + 0 * HID + 4);
    gf.q[0] = *(const float4*)(grow + 1 * HID);
    gf.q[1] = *(const float4*)(grow + 1 * HID + 4);
    gg.q[0] = *(const float4*)(grow + 2 * HID);
    gg.q[1] = *(const float4*)(grow + 2 * HID + 4);
    go.q[0] = *(const float4*)(grow + 3 * HID);
    go.q[1] = *(const float4*)(grow + 3 * HID + 4);

    float hv[8];
    float su0 = 0.f, su1 = 0.f, su2 = 0.f;
    float sp0 = 0.f, sp1 = 0.f, sp2 = 0.f;
#pragma unroll
    for (int jj = 0; jj < 8; ++jj) {
      float ii = fast_sig(gi.f[jj]);
      float ff = fast_sig(gf.f[jj]);
      float gz = fast_tanh(gg.f[jj]);
      float oo = fast_sig(go.f[jj]);
      c[jj] = ff * c[jj] + ii * gz;
      float h = oo * fast_tanh(c[jj]);
      hv[jj] = h;
      su0 += h * wu[0][jj]; su1 += h * wu[1][jj]; su2 += h * wu[2][jj];
      sp0 += h * wp[0][jj]; sp1 += h * wp[1][jj]; sp2 += h * wp[2][jj];
    }

    // h -> bf16 in LDS for next step's A-fragments
    unsigned hp0 = pk_bf16(hv[0], hv[1]);
    unsigned hp1 = pk_bf16(hv[2], hv[3]);
    unsigned hp2 = pk_bf16(hv[4], hv[5]);
    unsigned hp3 = pk_bf16(hv[6], hv[7]);
    *(uint4*)&h_lds[erow * HROW + (ej0 >> 1)] = make_uint4(hp0, hp1, hp2, hp3);

    // OUT-head reduction across the 32 threads sharing this row
    atomicAdd(&out_stage[erow][0], su0);
    atomicAdd(&out_stage[erow][1], su1);
    atomicAdd(&out_stage[erow][2], su2);
    atomicAdd(&out_stage[erow][3], sp0);
    atomicAdd(&out_stage[erow][4], sp1);
    atomicAdd(&out_stage[erow][5], sp2);

#if HAVE_TDM
    // make sure the prefetched x[t+1] tile has landed before publishing it
    if (tid == 0) __builtin_amdgcn_s_wait_tensorcnt((short)0);
#endif
    __syncthreads();

    // ---- write out[t, b0+row, k] and reset staging for the next step
    if (tid < BTILE * OUTD) {
      int row = tid / OUTD;
      int k   = tid % OUTD;
      float v = out_stage[row][k] + ((k < 3) ? buvw[k] : bpqr[k - 3]);
      out[((size_t)t * BATCH + b0 + row) * OUTD + k] = v;
      out_stage[row][k] = 0.0f;
    }
  }
}

extern "C" void kernel_launch(void* const* d_in, const int* in_sizes, int n_in,
                              void* d_out, int out_size, void* d_ws, size_t ws_size,
                              hipStream_t stream) {
  (void)in_sizes; (void)n_in; (void)out_size; (void)ws_size;
  const float* xs   = (const float*)d_in[0];
  const float* Wih  = (const float*)d_in[1];
  const float* Whh  = (const float*)d_in[2];
  const float* bih  = (const float*)d_in[3];
  const float* bhh  = (const float*)d_in[4];
  const float* Wuvw = (const float*)d_in[5];
  const float* buvw = (const float*)d_in[6];
  const float* Wpqr = (const float*)d_in[7];
  const float* bpqr = (const float*)d_in[8];
  float* out = (float*)d_out;

  unsigned* whh_pk = (unsigned*)d_ws;            // 512 KB
  unsigned* wih_pk = whh_pk + WHH_PK_U32;        // +64 KB

  int total = WHH_PK_U32 + WIH_PK_U32;
  pack_weights<<<(total + 255) / 256, 256, 0, stream>>>(Wih, Whh, whh_pk, wih_pk);
  lstm_scan<<<NBLOCKS, NTHREADS, 0, stream>>>(xs, whh_pk, wih_pk, bih, bhh,
                                              Wuvw, buvw, Wpqr, bpqr, out);
}
